// TestModel_15831249453215
// MI455X (gfx1250) — 8-bit matrix operand order repaired and checked by reading, compiled, not run on hardware
//
#include <hip/hip_runtime.h>


#define Hdim 4096
#define Tdim 8192

typedef __attribute__((ext_vector_type(16))) int   v16i;
typedef __attribute__((ext_vector_type(8)))  float v8f;
typedef __attribute__((ext_vector_type(4)))  unsigned v4u;
typedef __attribute__((ext_vector_type(8)))  int   v8i;
typedef __attribute__((ext_vector_type(4)))  int   v4i;
typedef unsigned long long u64;

#define USE_TDM __has_builtin(__builtin_amdgcn_tensor_load_to_lds)


__device__ __forceinline__ unsigned char f32_to_e4m3_sw(float x) {
  unsigned u = __float_as_uint(x);
  unsigned sign = (u >> 31) << 7;
  float a = fabsf(x);
  if (!(a > 0.0f)) return (unsigned char)sign;
  if (a >= 448.0f) return (unsigned char)(sign | 0x7E);
  if (a < 0.015625f) {
    int m = (int)rintf(a * 512.0f);
    if (m >= 8) return (unsigned char)(sign | 0x08);
    return (unsigned char)(sign | (unsigned)m);
  }
  unsigned au = __float_as_uint(a);
  au += 0x0007FFFF + ((au >> 20) & 1);
  int e = (int)(au >> 23) - 127 + 7;
  unsigned m = (au >> 20) & 7;
  if (e >= 16) return (unsigned char)(sign | 0x7E);
  if (e <= 0)  return (unsigned char)(sign | 0x08);
  return (unsigned char)(sign | ((unsigned)e << 4) | m);
}

__device__ __forceinline__ unsigned pack4_fp8(float a, float b, float c, float d) {
#if __has_builtin(__builtin_amdgcn_cvt_pk_fp8_f32)
  int v = __builtin_amdgcn_cvt_pk_fp8_f32(a, b, 0, false);
  v     = __builtin_amdgcn_cvt_pk_fp8_f32(c, d, v, true);
  return (unsigned)v;
#else
  return (unsigned)f32_to_e4m3_sw(a) | ((unsigned)f32_to_e4m3_sw(b) << 8) |
         ((unsigned)f32_to_e4m3_sw(c) << 16) | ((unsigned)f32_to_e4m3_sw(d) << 24);
#endif
}

__device__ __forceinline__ unsigned char f32_to_fp8(float a) {
#if __has_builtin(__builtin_amdgcn_cvt_pk_fp8_f32)
  return (unsigned char)(__builtin_amdgcn_cvt_pk_fp8_f32(a, 0.0f, 0, false) & 0xff);
#else
  return f32_to_e4m3_sw(a);
#endif
}


#if USE_TDM
__device__ __forceinline__ void tdm_load_tile_128x128(unsigned lds_addr,
                                                      const unsigned char* gptr,
                                                      unsigned tensor_rows) {
  const unsigned long long ga = (unsigned long long)(size_t)gptr;
  v4u g0;
  g0[0] = 1u;
  g0[1] = lds_addr;
  g0[2] = (unsigned)ga;
  g0[3] = (unsigned)((ga >> 32) & 0x01FFFFFFu) | (2u << 30);
  v8i g1;
  g1[0] = 0;
  g1[1] = (int)(4096u << 16);
  g1[2] = (int)(tensor_rows << 16);
  g1[3] = (int)(128u << 16);
  g1[4] = 128;
  g1[5] = 4096;
  g1[6] = 0;
  g1[7] = 0;
  v4i gz = {0, 0, 0, 0};
#if defined(__clang_major__) && (__clang_major__ >= 23)
  v8i gz8 = {0, 0, 0, 0, 0, 0, 0, 0};
  __builtin_amdgcn_tensor_load_to_lds(g0, g1, gz, gz, gz8, 0);
#else
  __builtin_amdgcn_tensor_load_to_lds(g0, g1, gz, gz, 0);
#endif
}
#endif

__device__ __forceinline__ void wait_tensorcnt_le2() {
#if __has_builtin(__builtin_amdgcn_s_wait_tensorcnt)
  __builtin_amdgcn_s_wait_tensorcnt(2);
#else
  asm volatile("s_wait_tensorcnt 0x2" ::: "memory");
#endif
}
__device__ __forceinline__ void wait_tensorcnt_0() {
#if __has_builtin(__builtin_amdgcn_s_wait_tensorcnt)
  __builtin_amdgcn_s_wait_tensorcnt(0);
#else
  asm volatile("s_wait_tensorcnt 0x0" ::: "memory");
#endif
}


__device__ __forceinline__ float block_reduce_sum(float v, float* sbuf) {
#pragma unroll
  for (int o = 16; o > 0; o >>= 1) v += __shfl_xor(v, o, 32);
  if ((threadIdx.x & 31) == 0) sbuf[threadIdx.x >> 5] = v;
  __syncthreads();
  float r = sbuf[0];
#pragma unroll
  for (int i = 1; i < 8; i++) r += sbuf[i];
  __syncthreads();
  return r;
}

__device__ __forceinline__ float block_reduce_max(float v, float* sbuf) {
#pragma unroll
  for (int o = 16; o > 0; o >>= 1) v = fmaxf(v, __shfl_xor(v, o, 32));
  if ((threadIdx.x & 31) == 0) sbuf[threadIdx.x >> 5] = v;
  __syncthreads();
  float r = sbuf[0];
#pragma unroll
  for (int i = 1; i < 8; i++) r = fmaxf(r, sbuf[i]);
  __syncthreads();
  return r;
}


__global__ __launch_bounds__(256) void quant_w_kernel(const float* __restrict__ w,
                                                      unsigned char* __restrict__ wqT) {
  __shared__ float tile[32][33];
  const int h0 = blockIdx.x * 32, k0 = blockIdx.y * 32;
  const int tx = threadIdx.x, ty = threadIdx.y;
#pragma unroll
  for (int r = 0; r < 4; r++) {
    int h = ty + r * 8;
    tile[h][tx] = w[(size_t)(h0 + h) * Hdim + k0 + tx];
  }
  __syncthreads();
#pragma unroll
  for (int r = 0; r < 4; r++) {
    int k = ty + r * 8;
    wqT[(size_t)(k0 + k) * Hdim + h0 + tx] = f32_to_fp8(tile[tx][k]);
  }
}


__global__ __launch_bounds__(256) void rms_quant_kernel(const float* __restrict__ in,
                                                        const float* __restrict__ nw,
                                                        float* __restrict__ resid_out,
                                                        unsigned char* __restrict__ qbuf,
                                                        float* __restrict__ srow,
                                                        int do_sqrt) {
  __shared__ float sbuf[8];
  const int tid = threadIdx.x;
  const int row = blockIdx.x;
  const float* xr = in + (size_t)row * Hdim;

  float4 v[4];
  float ss = 0.0f;
#pragma unroll
  for (int i = 0; i < 4; i++) {
    v[i] = *(const float4*)(xr + i * 1024 + tid * 4);
    ss += v[i].x * v[i].x + v[i].y * v[i].y + v[i].z * v[i].z + v[i].w * v[i].w;
  }
  if (do_sqrt) {
#pragma unroll
    for (int i = 0; i < 4; i++) {
      float4 s4 = {sqrtf(v[i].x), sqrtf(v[i].y), sqrtf(v[i].z), sqrtf(v[i].w)};
      *(float4*)(resid_out + (size_t)row * Hdim + i * 1024 + tid * 4) = s4;
    }
  }
  ss = block_reduce_sum(ss, sbuf);
  const float r = rsqrtf(ss * (1.0f / Hdim) + 1e-6f);

  float4 y[4];
  float mx = 0.0f;
#pragma unroll
  for (int i = 0; i < 4; i++) {
    float4 w4 = *(const float4*)(nw + i * 1024 + tid * 4);
    y[i].x = v[i].x * r * w4.x;
    y[i].y = v[i].y * r * w4.y;
    y[i].z = v[i].z * r * w4.z;
    y[i].w = v[i].w * r * w4.w;
    mx = fmaxf(mx, fmaxf(fmaxf(fabsf(y[i].x), fabsf(y[i].y)),
                         fmaxf(fabsf(y[i].z), fabsf(y[i].w))));
  }
  mx = block_reduce_max(mx, sbuf);
  const float s = fmaxf(mx * (1.0f / 448.0f), 1e-12f);
  if (tid == 0) srow[row] = s;
  const float inv = 1.0f / s;
#pragma unroll
  for (int i = 0; i < 4; i++) {
    unsigned p = pack4_fp8(y[i].x * inv, y[i].y * inv, y[i].z * inv, y[i].w * inv);
    *(unsigned*)(qbuf + (size_t)row * Hdim + i * 1024 + tid * 4) = p;
  }
}


__global__ __launch_bounds__(256) void gemm_fp8_kernel(const unsigned char* __restrict__ Aq,
                                                       const unsigned char* __restrict__ BqT,
                                                       const float* __restrict__ srow,
                                                       const float* __restrict__ wscale,
                                                       float* __restrict__ resid) {
#if USE_TDM
  __shared__ __attribute__((aligned(16))) unsigned char sA[2][128 * 128];
  __shared__ __attribute__((aligned(16))) unsigned char sB[2][128 * 128];
#else
  __shared__ __attribute__((aligned(16))) unsigned char sA[1][128 * 128];
  __shared__ __attribute__((aligned(16))) unsigned char sB[1][128 * 128];
#endif

  const int tid  = threadIdx.x;
  const int lane = tid & 31;
  const int wave = tid >> 5;
  const int wm   = wave >> 2;
  const int wn   = wave & 3;
  const int t0   = blockIdx.y * 128;
  const int n0   = blockIdx.x * 128;

  const v8f vzero = {0.f, 0.f, 0.f, 0.f, 0.f, 0.f, 0.f, 0.f};
  v8f acc[4][2];
#pragma unroll
  for (int mi = 0; mi < 4; mi++)
#pragma unroll
    for (int ni = 0; ni < 2; ni++) acc[mi][ni] = vzero;

  const int mrow_lo = lane & 15;
  const int a_hi = (lane & 16) ? 8 : 0;
  const int b_hi = (lane & 16) ? 8 : 0;

  const unsigned char* gA = Aq + (size_t)t0 * Hdim;
  const unsigned char* gB = BqT + (size_t)n0 * Hdim;

#if USE_TDM
  if (wave == 0) {
    tdm_load_tile_128x128((unsigned)(size_t)&sA[0][0], gA, Tdim);
    tdm_load_tile_128x128((unsigned)(size_t)&sB[0][0], gB, Hdim);
  }
#else
  const int ldr = tid >> 1;
  const int ldo = (tid & 1) * 64;
#endif

  for (int kk = 0; kk < Hdim / 128; kk++) {
#if USE_TDM
    const int cur = kk & 1;
    if (wave == 0) {
      if (kk + 1 < Hdim / 128) {
        tdm_load_tile_128x128((unsigned)(size_t)&sA[cur ^ 1][0], gA + (kk + 1) * 128, Tdim);
        tdm_load_tile_128x128((unsigned)(size_t)&sB[cur ^ 1][0], gB + (kk + 1) * 128, Hdim);
        wait_tensorcnt_le2();
      } else {
        wait_tensorcnt_0();
      }
      asm volatile("" ::: "memory");
    }
    __syncthreads();
    const unsigned char* tA = &sA[cur][0];
    const unsigned char* tB = &sB[cur][0];
#else
    const int h0 = kk * 128;
    {
      const uint4* ga = (const uint4*)(gA + (size_t)ldr * Hdim + h0 + ldo);
      uint4* la = (uint4*)(&sA[0][0] + ldr * 128 + ldo);
      la[0] = ga[0]; la[1] = ga[1]; la[2] = ga[2]; la[3] = ga[3];
      const uint4* gb = (const uint4*)(gB + (size_t)ldr * Hdim + h0 + ldo);
      uint4* lb = (uint4*)(&sB[0][0] + ldr * 128 + ldo);
      lb[0] = gb[0]; lb[1] = gb[1]; lb[2] = gb[2]; lb[3] = gb[3];
    }
    __syncthreads();
    const unsigned char* tA = &sA[0][0];
    const unsigned char* tB = &sB[0][0];
#endif

    v16i afr[4];
#pragma unroll
    for (int mi = 0; mi < 4; mi++) {
      const unsigned char* ap = tA + (wm * 64 + mi * 16 + mrow_lo) * 128;
#pragma unroll
      for (int c = 0; c < 8; c++) {
        u64 d = *(const u64*)(ap + c * 16 + a_hi);
        afr[mi][2 * c]     = (int)(unsigned)d;
        afr[mi][2 * c + 1] = (int)(d >> 32);
      }
    }
    v16i bfr[2];
#pragma unroll
    for (int ni = 0; ni < 2; ni++) {
      const unsigned char* bp = tB + (wn * 32 + ni * 16 + mrow_lo) * 128;
#pragma unroll
      for (int j = 0; j < 4; j++) {
        const u64 d0 = *(const u64*)(bp + j * 32 + b_hi);
        const u64 d1 = *(const u64*)(bp + j * 32 + 16 + b_hi);
        bfr[ni][4 * j + 0] = (int)(unsigned)d0;
        bfr[ni][4 * j + 1] = (int)(d0 >> 32);
        bfr[ni][4 * j + 2] = (int)(unsigned)d1;
        bfr[ni][4 * j + 3] = (int)(d1 >> 32);
      }
    }
#pragma unroll
    for (int mi = 0; mi < 4; mi++)
#pragma unroll
      for (int ni = 0; ni < 2; ni++)
        acc[mi][ni] = __builtin_amdgcn_wmma_f32_16x16x128_fp8_fp8(
            afr[mi], bfr[ni], (short)0, acc[mi][ni], false, false);
    __syncthreads();
  }

  const float wsc = wscale[0];
#pragma unroll
  for (int mi = 0; mi < 4; mi++) {
    const int mbase = t0 + wm * 64 + mi * 16 + ((lane & 16) ? 8 : 0);
#pragma unroll
    for (int ni = 0; ni < 2; ni++) {
      const int ncol = n0 + wn * 32 + ni * 16 + (lane & 15);
#pragma unroll
      for (int j = 0; j < 8; j++) {
        const int mrow = mbase + j;
        float* p = resid + (size_t)mrow * Hdim + ncol;
        *p += acc[mi][ni][j] * (srow[mrow] * wsc);
      }
    }
  }
}


__global__ __launch_bounds__(256) void final_norm_kernel(float* __restrict__ io,
                                                         const float* __restrict__ nw) {
  __shared__ float sbuf[8];
  const int tid = threadIdx.x;
  const int row = blockIdx.x;
  float* xr = io + (size_t)row * Hdim;
  float4 v[4];
  float ss = 0.0f;
#pragma unroll
  for (int i = 0; i < 4; i++) {
    v[i] = *(const float4*)(xr + i * 1024 + tid * 4);
    ss += v[i].x * v[i].x + v[i].y * v[i].y + v[i].z * v[i].z + v[i].w * v[i].w;
  }
  ss = block_reduce_sum(ss, sbuf);
  const float r = rsqrtf(ss * (1.0f / Hdim) + 1e-6f);
#pragma unroll
  for (int i = 0; i < 4; i++) {
    float4 w4 = *(const float4*)(nw + i * 1024 + tid * 4);
    float4 o = {v[i].x * r * w4.x, v[i].y * r * w4.y, v[i].z * r * w4.z, v[i].w * r * w4.w};
    *(float4*)(xr + i * 1024 + tid * 4) = o;
  }
}


extern "C" void kernel_launch(void* const* d_in, const int* in_sizes, int n_in,
                              void* d_out, int out_size, void* d_ws, size_t ws_size,
                              hipStream_t stream) {
  const float* x       = (const float*)d_in[0];
  const float* nw0     = (const float*)d_in[1];
  const float* nw1     = (const float*)d_in[2];
  const float* nw2     = (const float*)d_in[3];
  const float* w0      = (const float*)d_in[4];
  const float* w1      = (const float*)d_in[5];
  const float* wscale0 = (const float*)d_in[6];
  const float* wscale1 = (const float*)d_in[7];
  float* out = (float*)d_out;

  unsigned char* ws  = (unsigned char*)d_ws;
  unsigned char* w0q = ws;
  unsigned char* w1q = w0q + (size_t)Hdim * Hdim;
  unsigned char* qb  = w1q + (size_t)Hdim * Hdim;
  float* srow        = (float*)(qb + (size_t)Tdim * Hdim);

  dim3 bq(32, 8), gq(Hdim / 32, Hdim / 32);
  quant_w_kernel<<<gq, bq, 0, stream>>>(w0, w0q);
  quant_w_kernel<<<gq, bq, 0, stream>>>(w1, w1q);

  dim3 gg(Hdim / 128, Tdim / 128);

  rms_quant_kernel<<<Tdim, 256, 0, stream>>>(x, nw0, out, qb, srow, 1);
  gemm_fp8_kernel<<<gg, 256, 0, stream>>>(qb, w0q, srow, wscale0, out);

  rms_quant_kernel<<<Tdim, 256, 0, stream>>>(out, nw1, nullptr, qb, srow, 0);
  gemm_fp8_kernel<<<gg, 256, 0, stream>>>(qb, w1q, srow, wscale1, out);

  final_norm_kernel<<<Tdim, 256, 0, stream>>>(out, nw2);
}
